// GRNN_66142496358624
// MI455X (gfx1250) — matrix operand order repaired and checked by reading, not compiled, not run on hardware
//
#include <hip/hip_runtime.h>
#include <hip/hip_bf16.h>
#include <math.h>

typedef __attribute__((ext_vector_type(16))) _Float16 v16h;
typedef __attribute__((ext_vector_type(8)))  float    v8f;

#define SENT 64
#define WRD  32
#define DIM  300
#define HID  50
#define OUTC 5
#define NPAD 64          // padded N for the word-linear GEMM
#define KCHUNKS 10       // 10*32 = 320 >= 300
#define MTILES 128       // 2048 rows / 16
#define NTILES 4         // 64 cols / 16

// Fragment workspace layout (halves): frag[(tile*KCHUNKS + kc)*32 + lane] is a
// contiguous v16h (32 B) holding exactly the halves that lane feeds to WMMA.

// ---------------------------------------------------------------------------
// Prep: gather emb[doc] -> A fragments (f16, WMMA A-layout), wl_w -> B frags.
// A order used here (16-bit A 16x32, contiguous k, not the ISA table order): lane<16 row=lane K=0..15; lane>=16
// row=lane-16 K=16..31.  B order (same contiguous k as A): lane%16=col, lanes<16 K=0..15,
// lanes>=16 K=16..31, two K per VGPR.
// ---------------------------------------------------------------------------
__global__ __launch_bounds__(256)
void k_prep_frags(const int* __restrict__ doc,
                  const float* __restrict__ emb,
                  const float* __restrict__ wl_w,
                  _Float16* __restrict__ Afrag,
                  _Float16* __restrict__ Bfrag)
{
    const int t = blockIdx.x * 256 + threadIdx.x;
    const int A_TASKS = MTILES * KCHUNKS * 32;           // 40960

    _Float16 h[16];

    if (t < A_TASKS) {
        const int mtile = t / (KCHUNKS * 32);
        const int rem   = t % (KCHUNKS * 32);
        const int kc    = rem >> 5;
        const int lane  = rem & 31;
        const int row   = lane & 15;
        const int tok   = doc[mtile * 16 + row];
        const float* rowp = emb + (size_t)tok * DIM;
        const int ak    = (lane < 16) ? 0 : 16;
        const int base0 = kc * 32 + ak;
        #pragma unroll
        for (int g = 0; g < 2; ++g) {
            const int kk0 = base0 + g * 8;
            #pragma unroll
            for (int j = 0; j < 8; ++j) {
                const int kk = kk0 + j;
                h[g * 8 + j] = (_Float16)((kk < DIM) ? rowp[kk] : 0.f);
            }
        }
        uint4* dst = (uint4*)(Afrag + (size_t)t * 16);
        dst[0] = *(uint4*)&h[0];
        dst[1] = *(uint4*)&h[8];
    } else if (t < A_TASKS + NTILES * KCHUNKS * 32) {
        const int u    = t - A_TASKS;
        const int wid  = u / (KCHUNKS * 32);
        const int rem  = u % (KCHUNKS * 32);
        const int kc   = rem >> 5;
        const int lane = rem & 31;
        const int bcol = (lane & 15) + wid * 16;
        const int bk   = (lane < 16) ? 0 : 16;
        #pragma unroll
        for (int i = 0; i < 16; ++i) {
            const int kk = kc * 32 + bk + i;
            const float v = (bcol < HID && kk < DIM) ? wl_w[bcol * DIM + kk] : 0.f;
            h[i] = (_Float16)v;
        }
        uint4* dst = (uint4*)(Bfrag + (size_t)u * 16);
        dst[0] = *(uint4*)&h[0];
        dst[1] = *(uint4*)&h[8];
    }
}

// ---------------------------------------------------------------------------
// GEMM: one wave per 16x16 output tile. No LDS, no syncs; 10 unrolled WMMAs
// fed by aligned contiguous b128 loads from the fragment buffers.
// P[m][n] = sum_k emb[doc[m]][k]*wl_w[n][k] + wl_b[n].
// ---------------------------------------------------------------------------
__global__ __launch_bounds__(32)
void k_wordlinear_wmma(const _Float16* __restrict__ Afrag,
                       const _Float16* __restrict__ Bfrag,
                       const float* __restrict__ wl_b,
                       float* __restrict__ P)
{
    const int mtile = blockIdx.x >> 2;
    const int wid   = blockIdx.x & 3;
    const int lane  = threadIdx.x;

    const v16h* Af = (const v16h*)Afrag;
    const v16h* Bf = (const v16h*)Bfrag;

    v8f acc = {};
    #pragma unroll
    for (int kc = 0; kc < KCHUNKS; ++kc) {
        const v16h a = Af[(size_t)(mtile * KCHUNKS + kc) * 32 + lane];
        const v16h b = Bf[(size_t)(wid   * KCHUNKS + kc) * 32 + lane];
        acc = __builtin_amdgcn_wmma_f32_16x16x32_f16(
                  false, a, false, b, (short)0, acc, false, false);
    }

    // C/D layout: VGPR v -> M = v + (lane<16 ? 0 : 8), N = lane%16 (+ tile)
    const int rbase = (lane < 16) ? 0 : 8;
    const int col   = (lane & 15) + wid * 16;
    const float bias = (col < HID) ? wl_b[col] : 0.f;
    #pragma unroll
    for (int v = 0; v < 8; ++v) {
        const int m = mtile * 16 + rbase + v;
        P[(size_t)m * NPAD + col] = acc[v] + bias;
    }
}

// ---------------------------------------------------------------------------
// Conv branches collapsed to edge-corrected sums:
// mean_t conv(x)[o,t] = (1/L) * sum_{i,k} w[o,i,k] * Q[i,k],
// Q[i,k] = T[i] minus edge words.
// ---------------------------------------------------------------------------
__global__ __launch_bounds__(64)
void k_conv_rep(const float* __restrict__ P,
                const float* __restrict__ c1_w, const float* __restrict__ c1_b,
                const float* __restrict__ c2_w, const float* __restrict__ c2_b,
                const float* __restrict__ c3_w, const float* __restrict__ c3_b,
                float* __restrict__ rep)
{
    const int s = blockIdx.x;
    const int t = threadIdx.x;
    __shared__ float T[HID], e0[HID], e1[HID], e30[HID], e31[HID];

    if (t < HID) {
        const float* ps = P + (size_t)s * WRD * NPAD;
        float sum = 0.f;
        #pragma unroll
        for (int w = 0; w < WRD; ++w) sum += ps[w * NPAD + t];
        T[t]   = sum;
        e0[t]  = ps[0 * NPAD + t];
        e1[t]  = ps[1 * NPAD + t];
        e30[t] = ps[30 * NPAD + t];
        e31[t] = ps[31 * NPAD + t];
    }
    __syncthreads();

    if (t < HID) {
        float z1 = 0.f, z2 = 0.f, z3 = 0.f;
        for (int hi = 0; hi < HID; ++hi) {
            const float Tv = T[hi];
            z1 += c1_w[t * HID + hi] * Tv;                                  // K=1: full sum
            z2 += c2_w[(t * HID + hi) * 2 + 0] * (Tv - e31[hi])             // K=2,k=0
                + c2_w[(t * HID + hi) * 2 + 1] * (Tv - e0[hi]);             // K=2,k=1
            z3 += c3_w[(t * HID + hi) * 3 + 0] * (Tv - e30[hi] - e31[hi])   // K=3,k=0
                + c3_w[(t * HID + hi) * 3 + 1] * (Tv - e0[hi]  - e31[hi])   // K=3,k=1
                + c3_w[(t * HID + hi) * 3 + 2] * (Tv - e0[hi]  - e1[hi]);   // K=3,k=2
        }
        float r1 = tanhf(z1 * (1.f / 32.f) + c1_b[t]);
        float r2 = tanhf(z2 * (1.f / 31.f) + c2_b[t]);
        float r3 = tanhf(z3 * (1.f / 30.f) + c3_b[t]);
        rep[s * HID + t] = (r1 + r2 + r3) * (1.f / 3.f);
    }
}

// ---------------------------------------------------------------------------
// Bidirectional gated scan + mean + output layer + softmax (single block).
// Threads 0..49 forward, 64..113 backward.
// ---------------------------------------------------------------------------
__global__ __launch_bounds__(128)
void k_gnn_out(const float* __restrict__ rep,
               const float* __restrict__ fiw, const float* __restrict__ fib,
               const float* __restrict__ ffw, const float* __restrict__ ffb,
               const float* __restrict__ fgw, const float* __restrict__ fgb,
               const float* __restrict__ biw, const float* __restrict__ bib,
               const float* __restrict__ bfw, const float* __restrict__ bfb,
               const float* __restrict__ bgw, const float* __restrict__ bgb,
               const float* __restrict__ out_w, const float* __restrict__ out_b,
               float* __restrict__ out)
{
    __shared__ float reps[SENT][HID];
    __shared__ float hf[HID], hb[HID], sf[HID], sb[HID];

    const int t = threadIdx.x;
    for (int i = t; i < SENT * HID; i += 128) reps[i / HID][i % HID] = rep[i];
    if (t < HID) { hf[t] = 0.f; hb[t] = 0.f; sf[t] = 0.f; sb[t] = 0.f; }
    __syncthreads();

    const bool fwd = (t < HID);
    const bool bwd = (t >= 64 && t < 64 + HID);
    const int  ho  = fwd ? t : (t - 64);
    const float* Wi = fwd ? fiw : biw;
    const float* Wf = fwd ? ffw : bfw;
    const float* Wg = fwd ? fgw : bgw;
    const float* Bi = fwd ? fib : bib;
    const float* Bf = fwd ? ffb : bfb;
    const float* Bg = fwd ? fgb : bgb;

    for (int step = 0; step < SENT; ++step) {
        float hnew = 0.f;
        if (fwd || bwd) {
            const float* r = fwd ? reps[step] : reps[SENT - 1 - step];
            const float* h = fwd ? hf : hb;
            float zi = Bi[ho], zf = Bf[ho], zg = Bg[ho];
            for (int j = 0; j < HID; ++j) {
                const float rv = r[j], hv = h[j];
                zi += Wi[ho * 2 * HID + j] * rv + Wi[ho * 2 * HID + HID + j] * hv;
                zf += Wf[ho * 2 * HID + j] * rv + Wf[ho * 2 * HID + HID + j] * hv;
                zg += Wg[ho * 2 * HID + j] * rv + Wg[ho * 2 * HID + HID + j] * hv;
            }
            const float gi = 1.f / (1.f + expf(-zi));
            const float gf = 1.f / (1.f + expf(-zf));
            const float gg = tanhf(zg);
            hnew = tanhf(gi * gg + gf * (fwd ? hf[ho] : hb[ho]));
        }
        __syncthreads();
        if (fwd) { hf[ho] = hnew; sf[ho] += hnew; }
        if (bwd) { hb[ho] = hnew; sb[ho] += hnew; }
        __syncthreads();
    }

    if (t == 0) {
        float logits[OUTC];
        float maxv = -1e30f;
        for (int o = 0; o < OUTC; ++o) {
            float z = out_b[o];
            for (int j = 0; j < HID; ++j) z += out_w[o * 2 * HID + j]       * (sf[j] / SENT);
            for (int j = 0; j < HID; ++j) z += out_w[o * 2 * HID + HID + j] * (sb[j] / SENT);
            logits[o] = z;
            maxv = fmaxf(maxv, z);
        }
        float den = 0.f;
        for (int o = 0; o < OUTC; ++o) { logits[o] = expf(logits[o] - maxv); den += logits[o]; }
        for (int o = 0; o < OUTC; ++o) out[o] = logits[o] / den;
    }
}

// ---------------------------------------------------------------------------
extern "C" void kernel_launch(void* const* d_in, const int* in_sizes, int n_in,
                              void* d_out, int out_size, void* d_ws, size_t ws_size,
                              hipStream_t stream)
{
    const int*   doc   = (const int*)  d_in[0];
    const float* emb   = (const float*)d_in[1];
    const float* wl_w  = (const float*)d_in[2];
    const float* wl_b  = (const float*)d_in[3];
    const float* c1_w  = (const float*)d_in[4];
    const float* c1_b  = (const float*)d_in[5];
    const float* c2_w  = (const float*)d_in[6];
    const float* c2_b  = (const float*)d_in[7];
    const float* c3_w  = (const float*)d_in[8];
    const float* c3_b  = (const float*)d_in[9];
    const float* f_i_w = (const float*)d_in[10];
    const float* f_i_b = (const float*)d_in[11];
    const float* f_f_w = (const float*)d_in[12];
    const float* f_f_b = (const float*)d_in[13];
    const float* f_g_w = (const float*)d_in[14];
    const float* f_g_b = (const float*)d_in[15];
    const float* b_i_w = (const float*)d_in[16];
    const float* b_i_b = (const float*)d_in[17];
    const float* b_f_w = (const float*)d_in[18];
    const float* b_f_b = (const float*)d_in[19];
    const float* b_g_w = (const float*)d_in[20];
    const float* b_g_b = (const float*)d_in[21];
    const float* out_w = (const float*)d_in[22];
    const float* out_b = (const float*)d_in[23];
    float* out = (float*)d_out;

    // Workspace layout (bytes, all 256-aligned):
    //   P     : 2048*64 f32            = 524288
    //   rep   : 64*50  f32             =  12800
    //   Afrag : 128*10*32*16 f16       = 1310720
    //   Bfrag : 4*10*32*16 f16         =  40960
    char* ws = (char*)d_ws;
    float*    P     = (float*)   (ws);
    float*    rep   = (float*)   (ws + 524288);
    _Float16* Afrag = (_Float16*)(ws + 524288 + 12800 + 256);   // 537344, 512-aligned? keep 16B align
    _Float16* Bfrag = (_Float16*)(ws + 524288 + 12800 + 256 + 1310720);

    const int A_TASKS = MTILES * KCHUNKS * 32;      // 40960
    const int B_TASKS = NTILES * KCHUNKS * 32;      // 1280
    const int prep_blocks = (A_TASKS + B_TASKS + 255) / 256;  // 165

    k_prep_frags<<<prep_blocks, 256, 0, stream>>>(doc, emb, wl_w, Afrag, Bfrag);
    k_wordlinear_wmma<<<MTILES * NTILES, 32, 0, stream>>>(Afrag, Bfrag, wl_b, P);
    k_conv_rep<<<SENT, 64, 0, stream>>>(P, c1_w, c1_b, c2_w, c2_b, c3_w, c3_b, rep);
    k_gnn_out<<<1, 128, 0, stream>>>(rep,
                                     f_i_w, f_i_b, f_f_w, f_f_b, f_g_w, f_g_b,
                                     b_i_w, b_i_b, b_f_w, b_f_b, b_g_w, b_g_b,
                                     out_w, out_b, out);
}
